// mgn_GRU_D_29961691857384
// MI455X (gfx1250) — matrix operand order repaired and checked by reading, not compiled, not run on hardware
//
#include <hip/hip_runtime.h>
#include <hip/hip_bf16.h>

#define D_IN 32
#define HIDN 128
#define WPB 4   // waves (16-row tiles) per block

typedef _Float16 f16;
typedef _Float16 v16h __attribute__((ext_vector_type(16)));
typedef _Float16 v8h  __attribute__((ext_vector_type(8)));
typedef _Float16 v4h  __attribute__((ext_vector_type(4)));
typedef float    v8f  __attribute__((ext_vector_type(8)));

// fragment-major weight buffer layout (each fragment = 512 f16 = one 32x16 B tile)
#define FB_GX 0     // 2 frags  : W_gx^T  (K=32,  N=32)
#define FB_GH 2     // 8 frags  : W_gh^T  (K=32,  N=128)
#define FB_WZ 10    // 8
#define FB_VZ 18    // 8
#define FB_UZ 26    // 32       : (K=128, N=128), frag = kt*8+nt
#define FB_WR 58
#define FB_VR 66
#define FB_UR 74
#define FB_WH 106
#define FB_VH 114
#define FB_UH 122
#define N_FRAGS 154
#define FRAG_HALVES (N_FRAGS * 512)
#define RED_BLOCKS 128

__device__ __forceinline__ void lds_fence() {
  // wave-private LDS staging: DS ops are in-order per wave; this stops the
  // compiler from reordering across phases and drains DScnt.
  asm volatile("s_wait_dscnt 0" ::: "memory");
}

__device__ __forceinline__ float sigmoid_f(float x) { return 1.0f / (1.0f + __expf(-x)); }
__device__ __forceinline__ float tanh_f(float x)    { return 2.0f * sigmoid_f(2.0f * x) - 1.0f; }

__device__ __forceinline__ v16h cat8(v8h lo, v8h hi) {
  return __builtin_shufflevector(lo, hi, 0,1,2,3,4,5,6,7,8,9,10,11,12,13,14,15);
}

__device__ __forceinline__ v4h pack4(float a, float b, float c, float d) {
  v4h r; r[0] = (f16)a; r[1] = (f16)b; r[2] = (f16)c; r[3] = (f16)d; return r;
}

// A fragment (16xK f16 row-major tile in LDS), K-tile kt.
// ISA 16-bit A 16x32 layout: lane<16 holds M=lane, K={0..7, 16..23}; lane>=16 K={8..15, 24..31}
__device__ __forceinline__ v16h load_A(const f16* tile, int ldk, int kt, int lane) {
  int m = lane & 15;
  int off = (kt << 5) + ((lane >> 4) << 3);
  const f16* p = tile + m * ldk + off;
  v8h lo = *(const v8h*)p;
  v8h hi = *(const v8h*)(p + 16);
  return cat8(lo, hi);
}

// B fragment from fragment-major global buffer (32 contiguous bytes per lane)
__device__ __forceinline__ v16h load_B(const f16* frags, int fragId, int lane) {
  const f16* p = frags + ((size_t)fragId << 9) + (lane << 4);
  v8h lo = *(const v8h*)p;
  v8h hi = *(const v8h*)(p + 8);
  return cat8(lo, hi);
}

__device__ __forceinline__ v8f wmma_f16(v16h a, v16h b, v8f c) {
  return __builtin_amdgcn_wmma_f32_16x16x32_f16(false, a, false, b, (short)0, c, false, false);
}

// ---------------- mean_x two-pass deterministic reduction ----------------
__global__ void reduce1(const float* __restrict__ X, const float* __restrict__ M,
                        float* __restrict__ part, int nrows) {
  __shared__ float sx[256], sm[256];
  int col = threadIdx.x & 31, grp = threadIdx.x >> 5;
  float ax = 0.0f, am = 0.0f;
  for (int r = blockIdx.x * 8 + grp; r < nrows; r += gridDim.x * 8) {
    ax += X[r * D_IN + col];
    am += M[r * D_IN + col];
  }
  sx[threadIdx.x] = ax; sm[threadIdx.x] = am;
  __syncthreads();
  if (threadIdx.x < 32) {
    float tx = 0.0f, tm = 0.0f;
    for (int g = 0; g < 8; ++g) { tx += sx[g * 32 + threadIdx.x]; tm += sm[g * 32 + threadIdx.x]; }
    part[blockIdx.x * 64 + threadIdx.x] = tx;
    part[blockIdx.x * 64 + 32 + threadIdx.x] = tm;
  }
}

__global__ void reduce2(const float* __restrict__ part, float* __restrict__ mean,
                        int nblocks, float denom_add) {
  int t = threadIdx.x;
  if (t < 32) {
    float sxv = 0.0f, smv = 0.0f;
    for (int b = 0; b < nblocks; ++b) { sxv += part[b * 64 + t]; smv += part[b * 64 + 32 + t]; }
    mean[t] = sxv / (smv + denom_add);
  }
}

// ---------------- weight staging: f32 [K,N] (opt transposed) -> f16 fragment-major ----------------
__device__ void stage_mat(const float* __restrict__ src, f16* __restrict__ dst,
                          int K, int N, bool tr, int gt, int gs) {
  int ntile = N >> 4;
  int tot = K * N;
  for (int e = gt; e < tot; e += gs) {
    int frag = e >> 9, w = e & 511, lane = w >> 4, j = w & 15;
    int kt = frag / ntile, nt = frag - kt * ntile;
    int k = (kt << 5) + ((lane >> 4) << 3) + (j & 7) + ((j >> 3) << 4);   // lane-half h: K={8h..8h+7, 16+8h..16+8h+7}
    int n = (nt << 4) + (lane & 15);
    float v = tr ? src[n * K + k] : src[k * N + n];
    dst[e] = (f16)v;
  }
}

__global__ void stage_weights(const float* Wgx, const float* Wgh,
                              const float* Wz, const float* Vz, const float* Uz,
                              const float* Wr, const float* Vr, const float* Ur,
                              const float* Wh, const float* Vh, const float* Uh,
                              f16* fr) {
  int gt = blockIdx.x * blockDim.x + threadIdx.x;
  int gs = gridDim.x * blockDim.x;
  stage_mat(Wgx, fr + (FB_GX << 9), 32, 32,  true,  gt, gs);
  stage_mat(Wgh, fr + (FB_GH << 9), 32, 128, true,  gt, gs);
  stage_mat(Wz,  fr + (FB_WZ << 9), 32, 128, false, gt, gs);
  stage_mat(Vz,  fr + (FB_VZ << 9), 32, 128, false, gt, gs);
  stage_mat(Uz,  fr + (FB_UZ << 9), 128,128, false, gt, gs);
  stage_mat(Wr,  fr + (FB_WR << 9), 32, 128, false, gt, gs);
  stage_mat(Vr,  fr + (FB_VR << 9), 32, 128, false, gt, gs);
  stage_mat(Ur,  fr + (FB_UR << 9), 128,128, false, gt, gs);
  stage_mat(Wh,  fr + (FB_WH << 9), 32, 128, false, gt, gs);
  stage_mat(Vh,  fr + (FB_VH << 9), 32, 128, false, gt, gs);
  stage_mat(Uh,  fr + (FB_UH << 9), 128,128, false, gt, gs);
}

// ---------------- main GRU-D kernel: one wave per 16-row tile ----------------
struct WaveLds {
  f16 Aint[16 * 32];   // interval (A for gamma GEMMs)
  f16 Am[16 * 32];     // mask     (A for M@V)
  f16 Axh[16 * 32];    // X_hat    (A for X@W)
  f16 Ah[16 * 128];    // decayed hidden h (A for h@U), also h for epilogue
  f16 Arh[16 * 128];   // gx temp -> r -> r*h -> h_tilde temp
  f16 Zt[16 * 128];    // gamma_h temp -> z gate
  int idx[16];
};

__global__ __launch_bounds__(WPB * 32) void grud_main(
    const float* __restrict__ t_cur_p, const float* __restrict__ mgn_h,
    const float* __restrict__ X_obs,   const float* __restrict__ M_obs,
    const int*   __restrict__ i_obs,   const float* __restrict__ last_x,
    const float* __restrict__ last_t,  const f16*   __restrict__ frags,
    const float* __restrict__ mean_x,  const float* __restrict__ bz,
    const float* __restrict__ br,      const float* __restrict__ bh,
    float* __restrict__ out_h, float* __restrict__ out_x, float* __restrict__ out_t,
    int nobs) {
  __shared__ __align__(16) WaveLds wls[WPB];
  const int lane = threadIdx.x & 31;
  const int wave = threadIdx.x >> 5;
  const int tile = blockIdx.x * WPB + wave;
  const int ntiles = (nobs + 15) >> 4;
  if (tile >= ntiles) return;                 // uniform per wave; EXEC stays all-ones
  WaveLds& wl = wls[wave];
  const int row0 = tile << 4;
  const float t_now = t_cur_p[0];

  // phase 0: gather row indices
  if (lane < 16) {
    int r = row0 + lane;
    wl.idx[lane] = i_obs[(r < nobs) ? r : (nobs - 1)];
  }
  lds_fence();

  // phase 1: stage interval & mask as f16 A tiles (vectorized, b128 loads / b64 ds stores)
  #pragma unroll
  for (int it = 0; it < 4; ++it) {
    int chunk = it * 32 + lane;        // 128 chunks of 4 cols over [16][32]
    int r = chunk >> 3, c4 = (chunk & 7) << 2;
    int rg = row0 + r; if (rg >= nobs) rg = nobs - 1;
    int i = wl.idx[r];
    float4 lt = *(const float4*)(last_t + (size_t)i * D_IN + c4);
    float4 m  = *(const float4*)(M_obs + (size_t)rg * D_IN + c4);
    *(v4h*)&wl.Aint[r * 32 + c4] = pack4(t_now - lt.x, t_now - lt.y, t_now - lt.z, t_now - lt.w);
    *(v4h*)&wl.Am[r * 32 + c4]   = pack4(m.x, m.y, m.z, m.w);
  }
  lds_fence();

  v16h a_int = load_A(wl.Aint, 32, 0, lane);
  const int n_lo = lane & 15;
  const int m_hi = (lane >> 4) << 3;

  // phase 2a: gamma_x -> f16 temp tile (reuse Arh as [16][32] gx scratch)
  f16* GxT = wl.Arh;
  #pragma unroll
  for (int nt = 0; nt < 2; ++nt) {
    v8f c = {};
    c = wmma_f16(a_int, load_B(frags, FB_GX + nt, lane), c);
    int cc = n_lo + (nt << 4);
    #pragma unroll
    for (int v = 0; v < 8; ++v) {
      int m = v + m_hi;
      GxT[m * 32 + cc] = (f16)__expf(-fmaxf(c[v], 0.0f));
    }
  }
  lds_fence();

  // phase 2b: vectorized scatter-update of last_x/last_t + X_hat staging
  #pragma unroll
  for (int it = 0; it < 4; ++it) {
    int chunk = it * 32 + lane;
    int r = chunk >> 3, c4 = (chunk & 7) << 2;
    int rg = row0 + r;
    bool ok = rg < nobs;
    int rgs = ok ? rg : nobs - 1;
    int i = wl.idx[r];
    float4 X  = *(const float4*)(X_obs + (size_t)rgs * D_IN + c4);
    float4 lx = *(const float4*)(last_x + (size_t)i * D_IN + c4);
    float4 lt = *(const float4*)(last_t + (size_t)i * D_IN + c4);
    float4 mn = *(const float4*)(mean_x + c4);
    v4h mh = *(const v4h*)&wl.Am[r * 32 + c4];
    v4h gh4 = *(const v4h*)&GxT[r * 32 + c4];
    float4 lxn, ltn;
    float xh[4];
    float* Xp = (float*)&X; float* lxp = (float*)&lx; float* ltp = (float*)&lt;
    float* mnp = (float*)&mn; float* lxnp = (float*)&lxn; float* ltnp = (float*)&ltn;
    #pragma unroll
    for (int q = 0; q < 4; ++q) {
      float M = (float)mh[q];
      float g = (float)gh4[q];
      float lxn_q = lxp[q] + M * (Xp[q] - lxp[q]);
      lxnp[q] = lxn_q;
      ltnp[q] = ltp[q] + M * (t_now - ltp[q]);
      xh[q] = M * Xp[q] + (1.0f - M) * (g * lxn_q + (1.0f - g) * mnp[q]);
    }
    if (ok) {
      *(float4*)(out_x + (size_t)i * D_IN + c4) = lxn;
      *(float4*)(out_t + (size_t)i * D_IN + c4) = ltn;
    }
    *(v4h*)&wl.Axh[r * 32 + c4] = pack4(xh[0], xh[1], xh[2], xh[3]);
  }

  // phase 3a: gamma_h -> f16 temp tile (reuse Zt as [16][128] gh scratch)
  #pragma unroll
  for (int nt = 0; nt < 8; ++nt) {
    v8f c = {};
    c = wmma_f16(a_int, load_B(frags, FB_GH + nt, lane), c);
    int cc = n_lo + (nt << 4);
    #pragma unroll
    for (int v = 0; v < 8; ++v) {
      int m = v + m_hi;
      wl.Zt[m * HIDN + cc] = (f16)__expf(-fmaxf(c[v], 0.0f));
    }
  }
  lds_fence();

  // phase 3b: h = gamma_h * mgn_h[idx]  (vectorized b128 gather)
  #pragma unroll
  for (int it = 0; it < 16; ++it) {
    int chunk = it * 32 + lane;        // 512 chunks of 4 cols over [16][128]
    int r = chunk >> 5, c4 = (chunk & 31) << 2;
    int i = wl.idx[r];
    float4 hv = *(const float4*)(mgn_h + (size_t)i * HIDN + c4);
    v4h g = *(const v4h*)&wl.Zt[r * HIDN + c4];
    *(v4h*)&wl.Ah[r * HIDN + c4] =
        pack4((float)g[0] * hv.x, (float)g[1] * hv.y, (float)g[2] * hv.z, (float)g[3] * hv.w);
  }
  lds_fence();

  v16h a_xh = load_A(wl.Axh, 32, 0, lane);
  v16h a_m  = load_A(wl.Am, 32, 0, lane);
  v16h a_h[4];
  #pragma unroll
  for (int kt = 0; kt < 4; ++kt) a_h[kt] = load_A(wl.Ah, 128, kt, lane);

  // phase 4a: r gate -> stage r into Arh
  for (int nt = 0; nt < 8; ++nt) {
    int cc = n_lo + (nt << 4);
    float b = br[cc];
    v8f c = {b, b, b, b, b, b, b, b};
    c = wmma_f16(a_xh, load_B(frags, FB_WR + nt, lane), c);
    c = wmma_f16(a_m,  load_B(frags, FB_VR + nt, lane), c);
    #pragma unroll
    for (int kt = 0; kt < 4; ++kt)
      c = wmma_f16(a_h[kt], load_B(frags, FB_UR + kt * 8 + nt, lane), c);
    #pragma unroll
    for (int v = 0; v < 8; ++v) {
      int m = v + m_hi;
      wl.Arh[m * HIDN + cc] = (f16)sigmoid_f(c[v]);
    }
  }
  lds_fence();

  // phase 4b: Arh = r * h (vectorized)
  #pragma unroll
  for (int it = 0; it < 16; ++it) {
    int chunk = it * 32 + lane;
    int r = chunk >> 5, c4 = (chunk & 31) << 2;
    v4h rv = *(const v4h*)&wl.Arh[r * HIDN + c4];
    v4h hv = *(const v4h*)&wl.Ah[r * HIDN + c4];
    *(v4h*)&wl.Arh[r * HIDN + c4] =
        pack4((float)rv[0] * (float)hv[0], (float)rv[1] * (float)hv[1],
              (float)rv[2] * (float)hv[2], (float)rv[3] * (float)hv[3]);
  }
  lds_fence();

  // phase 5: z gate -> Zt (overwrites gamma_h scratch)
  for (int nt = 0; nt < 8; ++nt) {
    int cc = n_lo + (nt << 4);
    float b = bz[cc];
    v8f c = {b, b, b, b, b, b, b, b};
    c = wmma_f16(a_xh, load_B(frags, FB_WZ + nt, lane), c);
    c = wmma_f16(a_m,  load_B(frags, FB_VZ + nt, lane), c);
    #pragma unroll
    for (int kt = 0; kt < 4; ++kt)
      c = wmma_f16(a_h[kt], load_B(frags, FB_UZ + kt * 8 + nt, lane), c);
    #pragma unroll
    for (int v = 0; v < 8; ++v) {
      int m = v + m_hi;
      wl.Zt[m * HIDN + cc] = (f16)sigmoid_f(c[v]);
    }
  }
  lds_fence();

  v16h a_rh[4];
  #pragma unroll
  for (int kt = 0; kt < 4; ++kt) a_rh[kt] = load_A(wl.Arh, 128, kt, lane);
  lds_fence();   // a_rh in registers; Arh now reusable as h_tilde scratch

  // phase 6a: h_tilde -> stage into Arh
  for (int nt = 0; nt < 8; ++nt) {
    int cc = n_lo + (nt << 4);
    float b = bh[cc];
    v8f c = {b, b, b, b, b, b, b, b};
    c = wmma_f16(a_xh, load_B(frags, FB_WH + nt, lane), c);
    c = wmma_f16(a_m,  load_B(frags, FB_VH + nt, lane), c);
    #pragma unroll
    for (int kt = 0; kt < 4; ++kt)
      c = wmma_f16(a_rh[kt], load_B(frags, FB_UH + kt * 8 + nt, lane), c);
    #pragma unroll
    for (int v = 0; v < 8; ++v) {
      int m = v + m_hi;
      wl.Arh[m * HIDN + cc] = (f16)tanh_f(c[v]);
    }
  }
  lds_fence();

  // phase 6b: h_new = h + z*(h_tilde - h), vectorized b128 scatter into out_h
  #pragma unroll
  for (int it = 0; it < 16; ++it) {
    int chunk = it * 32 + lane;
    int r = chunk >> 5, c4 = (chunk & 31) << 2;
    int rg = row0 + r;
    if (rg < nobs) {
      int i = wl.idx[r];
      v4h hv = *(const v4h*)&wl.Ah[r * HIDN + c4];
      v4h zv = *(const v4h*)&wl.Zt[r * HIDN + c4];
      v4h tv = *(const v4h*)&wl.Arh[r * HIDN + c4];
      float4 o;
      o.x = (float)hv[0] + (float)zv[0] * ((float)tv[0] - (float)hv[0]);
      o.y = (float)hv[1] + (float)zv[1] * ((float)tv[1] - (float)hv[1]);
      o.z = (float)hv[2] + (float)zv[2] * ((float)tv[2] - (float)hv[2]);
      o.w = (float)hv[3] + (float)zv[3] * ((float)tv[3] - (float)hv[3]);
      *(float4*)(out_h + (size_t)i * HIDN + c4) = o;
    }
  }
}

extern "C" void kernel_launch(void* const* d_in, const int* in_sizes, int n_in,
                              void* d_out, int out_size, void* d_ws, size_t ws_size,
                              hipStream_t stream) {
  const float* t_cur  = (const float*)d_in[0];
  const float* mgn_h  = (const float*)d_in[1];
  const float* X_obs  = (const float*)d_in[2];
  const float* M_obs  = (const float*)d_in[3];
  const int*   i_obs  = (const int*)d_in[4];
  const float* last_x = (const float*)d_in[5];
  const float* last_t = (const float*)d_in[6];
  const float* Wgx = (const float*)d_in[7];
  const float* Wgh = (const float*)d_in[8];
  const float* Wz = (const float*)d_in[9];
  const float* Vz = (const float*)d_in[10];
  const float* Uz = (const float*)d_in[11];
  const float* bz = (const float*)d_in[12];
  const float* Wr = (const float*)d_in[13];
  const float* Vr = (const float*)d_in[14];
  const float* Ur = (const float*)d_in[15];
  const float* br = (const float*)d_in[16];
  const float* Wh = (const float*)d_in[17];
  const float* Vh = (const float*)d_in[18];
  const float* Uh = (const float*)d_in[19];
  const float* bh = (const float*)d_in[20];

  const int N    = in_sizes[1] / HIDN;
  const int NOBS = in_sizes[2] / D_IN;

  float* out_h = (float*)d_out;
  float* out_x = out_h + (size_t)N * HIDN;
  float* out_t = out_x + (size_t)N * D_IN;

  f16* fr = (f16*)d_ws;
  size_t part_off = (((size_t)FRAG_HALVES * sizeof(f16)) + 255) & ~(size_t)255;
  float* part = (float*)((char*)d_ws + part_off);
  float* mean = part + RED_BLOCKS * 64;

  // default outputs = inputs; observed rows overwritten by grud_main
  hipMemcpyAsync(out_h, mgn_h,  (size_t)N * HIDN * sizeof(float), hipMemcpyDeviceToDevice, stream);
  hipMemcpyAsync(out_x, last_x, (size_t)N * D_IN * sizeof(float), hipMemcpyDeviceToDevice, stream);
  hipMemcpyAsync(out_t, last_t, (size_t)N * D_IN * sizeof(float), hipMemcpyDeviceToDevice, stream);

  reduce1<<<RED_BLOCKS, 256, 0, stream>>>(X_obs, M_obs, part, NOBS);
  reduce2<<<1, 32, 0, stream>>>(part, mean, RED_BLOCKS, 1e-6f * (float)NOBS);
  stage_weights<<<64, 256, 0, stream>>>(Wgx, Wgh, Wz, Vz, Uz, Wr, Vr, Ur, Wh, Vh, Uh, fr);

  int ntiles  = (NOBS + 15) / 16;
  int nblocks = (ntiles + WPB - 1) / WPB;
  grud_main<<<nblocks, WPB * 32, 0, stream>>>(t_cur, mgn_h, X_obs, M_obs, i_obs,
      last_x, last_t, fr, mean, bz, br, bh, out_h, out_x, out_t, NOBS);
}
